// PPSP_1up_8933531976323
// MI455X (gfx1250) — matrix operand order repaired and checked by reading, not compiled, not run on hardware
//
#include <hip/hip_runtime.h>

#define B_ROWS  32768
#define L_DIM   1024
#define HID     256
#define OUT_DIM 1024
#define KF      1365
#define KFP     1408   // K padded to multiple of 32
#define BM      64     // rows per workgroup
#define HSTRIDE 272    // hid LDS row stride (bf16 elems)

typedef __bf16  v16bf  __attribute__((ext_vector_type(16)));
typedef float   v8f    __attribute__((ext_vector_type(8)));
typedef unsigned int u4v __attribute__((ext_vector_type(4)));
typedef unsigned int u8v __attribute__((ext_vector_type(8)));

union Frag {
    v16bf bf;
    u8v   o;     // one 32-byte load
    u4v   q[2];  // two 16-byte loads
};

__device__ __forceinline__ unsigned short f2bf(float f) {
    unsigned int u = __float_as_uint(f);
    u += 0x7FFFu + ((u >> 16) & 1u);   // round-to-nearest-even
    return (unsigned short)(u >> 16);
}
__device__ __forceinline__ float bf2f(unsigned short s) {
    return __uint_as_float(((unsigned int)s) << 16);
}

// ---- Prep: W1 (256x1365 f32 -> 256x1408 bf16, zero-padded K) and
// ----       W2 (1024x256 f32 -> bf16) into workspace (L2-resident).
__global__ __launch_bounds__(256) void prep_weights(
    const float* __restrict__ W1, const float* __restrict__ W2,
    unsigned short* __restrict__ W1bf, unsigned short* __restrict__ W2bf)
{
    int idx    = blockIdx.x * 256 + threadIdx.x;
    int stride = gridDim.x * 256;
    for (int i = idx; i < HID * KFP; i += stride) {
        int r = i / KFP, k = i - r * KFP;
        W1bf[i] = (k < KF) ? f2bf(W1[r * KF + k]) : (unsigned short)0;
    }
    for (int i = idx; i < OUT_DIM * HID; i += stride) {
        W2bf[i] = f2bf(W2[i]);
    }
}

// ---- Fused: feat build (LDS) -> GEMM1+ReLU (LDS) -> GEMM2 -> fund; h copied out.
__global__ __launch_bounds__(256) void fused_mlp(
    const float* __restrict__ x,
    const unsigned short* __restrict__ W1bf,
    const float* __restrict__ b1,
    const unsigned short* __restrict__ W2bf,
    const float* __restrict__ b2,
    float* __restrict__ fund,
    float* __restrict__ hout)
{
    __shared__ __attribute__((aligned(16))) unsigned short featLDS[BM * KFP];     // 176 KB
    __shared__ __attribute__((aligned(16))) unsigned short hidLDS [BM * HSTRIDE]; // 34 KB

    const int tid  = threadIdx.x;
    const int lane = tid & 31;
    const int wave = tid >> 5;
    const size_t rowBase = (size_t)blockIdx.x * BM;

    // ---------------- Phase 1a: x tile -> featLDS (bf16) + h output (f32) ----
    {
        const float4* xv = (const float4*)(x    + rowBase * L_DIM);
        float4*       hv = (float4*)      (hout + rowBase * L_DIM);
        #pragma unroll 4
        for (int r = 0; r < BM; ++r) {           // one row per iteration, col4 = tid
            float4 v = xv[(size_t)r * 256 + tid];
            hv[(size_t)r * 256 + tid] = v;
            int base = r * KFP + tid * 4;
            featLDS[base + 0] = f2bf(v.x);
            featLDS[base + 1] = f2bf(v.y);
            featLDS[base + 2] = f2bf(v.z);
            featLDS[base + 3] = f2bf(v.w);
        }
        // zero the K padding (cols 1365..1407)
        for (int i = tid; i < BM * (KFP - KF); i += 256) {
            int r = i / (KFP - KF);
            int c = i - r * (KFP - KF);
            featLDS[r * KFP + KF + c] = 0;
        }
    }
    __syncthreads();

    // ---------------- Phase 1b: pooled features (cols 1024..1364) ------------
    // reads cols < 1024, writes cols >= 1024: no intra-phase hazard
    for (int i = tid; i < BM * 341; i += 256) {
        int r = i / 341;
        int j = i - r * 341;
        const unsigned short* row = featLDS + r * KFP;
        float s = bf2f(row[j])
                + 0.5f        * (bf2f(row[2*j]) + bf2f(row[2*j+1]))
                + (1.0f/3.0f) * (bf2f(row[3*j]) + bf2f(row[3*j+1]) + bf2f(row[3*j+2]));
        featLDS[r * KFP + L_DIM + j] = f2bf(s);
    }
    __syncthreads();

    // WMMA lane decomposition (per ISA 16-bit A/B and f32 C/D layouts)
    const int m      = lane & 15;
    const int aKoff  = (lane & 16) ? 8  : 0;  // A-matrix 16x32 bf16
    const int bKoff  = (lane & 16) ? 8  : 0;  // B-matrix 32x16 bf16
    const int rowOff = (lane & 16) ? 8  : 0;  // C/D

    // ---------------- Phase 2: GEMM1  hid = relu(feat @ W1^T + b1) -----------
    // Each wave: 4 row-blocks x 2 exclusive col-blocks -> W1 read ONCE per WG.
    {
        const int cb0 = wave * 2;             // exclusive col tiles {cb0, cb0+1}
        v8f acc[4][2];
        #pragma unroll
        for (int r = 0; r < 4; ++r)
            #pragma unroll
            for (int t = 0; t < 2; ++t) acc[r][t] = (v8f){};

        for (int k = 0; k < KFP; k += 32) {
            Frag a[4];
            #pragma unroll
            for (int r = 0; r < 4; ++r) {
                const unsigned short* ar = featLDS + (r * 16 + m) * KFP + k;
                a[r].q[0] = *(const u4v*)(ar + aKoff);
                a[r].q[1] = *(const u4v*)(ar + 16 + aKoff);
            }
            #pragma unroll
            for (int t = 0; t < 2; ++t) {
                int n = (cb0 + t) * 16 + m;
                Frag b;
                b.q[0] = *(const u4v*)(W1bf + (size_t)n * KFP + k + bKoff);
                b.q[1] = *(const u4v*)(W1bf + (size_t)n * KFP + k + 16 + bKoff);
                #pragma unroll
                for (int r = 0; r < 4; ++r)
                    acc[r][t] = __builtin_amdgcn_wmma_f32_16x16x32_bf16(
                        false, a[r].bf, false, b.bf, (short)0, acc[r][t], false, false);
            }
        }
        #pragma unroll
        for (int t = 0; t < 2; ++t) {
            int colL   = (cb0 + t) * 16 + m;
            float bias = b1[colL];
            #pragma unroll
            for (int r = 0; r < 4; ++r)
                #pragma unroll
                for (int j = 0; j < 8; ++j) {
                    float v = acc[r][t][j] + bias;
                    v = v > 0.0f ? v : 0.0f;
                    hidLDS[(r * 16 + j + rowOff) * HSTRIDE + colL] = f2bf(v);
                }
        }
    }
    __syncthreads();

    // ---------------- Phase 3: GEMM2  fund = hid @ W2^T + b2 -----------------
    // Each wave: 4 row-blocks x 8 exclusive col-blocks (four 4x2 passes)
    // -> W2 read ONCE per WG; A re-read from LDS per pass (cheap).
    {
        const int cbBase = wave * 8;
        for (int p = 0; p < 4; ++p) {
            v8f acc[4][2];
            #pragma unroll
            for (int r = 0; r < 4; ++r)
                #pragma unroll
                for (int t = 0; t < 2; ++t) acc[r][t] = (v8f){};

            #pragma unroll
            for (int kk = 0; kk < 8; ++kk) {
                Frag a[4];
                #pragma unroll
                for (int r = 0; r < 4; ++r) {
                    const unsigned short* ar = hidLDS + (r * 16 + m) * HSTRIDE + kk * 32;
                    a[r].q[0] = *(const u4v*)(ar + aKoff);
                    a[r].q[1] = *(const u4v*)(ar + 16 + aKoff);
                }
                #pragma unroll
                for (int t = 0; t < 2; ++t) {
                    int colG = (cbBase + 2 * p + t) * 16 + m;
                    Frag b;
                    b.q[0] = *(const u4v*)(W2bf + (size_t)colG * HID + kk * 32 + bKoff);
                    b.q[1] = *(const u4v*)(W2bf + (size_t)colG * HID + kk * 32 + 16 + bKoff);
                    #pragma unroll
                    for (int r = 0; r < 4; ++r)
                        acc[r][t] = __builtin_amdgcn_wmma_f32_16x16x32_bf16(
                            false, a[r].bf, false, b.bf, (short)0, acc[r][t], false, false);
                }
            }
            #pragma unroll
            for (int t = 0; t < 2; ++t) {
                int colG   = (cbBase + 2 * p + t) * 16 + m;
                float bias = b2[colG];
                #pragma unroll
                for (int r = 0; r < 4; ++r)
                    #pragma unroll
                    for (int j = 0; j < 8; ++j) {
                        size_t rg = rowBase + (size_t)(r * 16 + j + rowOff);
                        fund[rg * OUT_DIM + colG] = acc[r][t][j] + bias;
                    }
            }
        }
    }
}

extern "C" void kernel_launch(void* const* d_in, const int* in_sizes, int n_in,
                              void* d_out, int out_size, void* d_ws, size_t ws_size,
                              hipStream_t stream) {
    const float* x  = (const float*)d_in[0];
    const float* W1 = (const float*)d_in[1];
    const float* b1 = (const float*)d_in[2];
    const float* W2 = (const float*)d_in[3];
    const float* b2 = (const float*)d_in[4];

    float* fund = (float*)d_out;                            // B*1024 floats
    float* hout = (float*)d_out + (size_t)B_ROWS * OUT_DIM; // then B*1024 floats

    unsigned short* W1bf = (unsigned short*)d_ws;           // 256*1408 bf16
    unsigned short* W2bf = W1bf + (size_t)HID * KFP;        // 1024*256 bf16

    prep_weights<<<512, 256, 0, stream>>>(W1, W2, W1bf, W2bf);
    fused_mlp<<<B_ROWS / BM, 256, 0, stream>>>(x, W1bf, b1, W2bf, b2, fund, hout);
}
